// DFlashAttention_43774306681111
// MI455X (gfx1250) — matrix operand order repaired and checked by reading, not compiled, not run on hardware
//
#include <hip/hip_runtime.h>
#include <hip/hip_bf16.h>

// Problem constants (match reference)
#define B_C    2
#define S_C    2048
#define HID_C  2048
#define NH_C   16
#define NKV_C  4
#define D_C    128
#define TOK_C  (B_C * S_C)            // 4096
#define SCALE_C 0.08838834764831845f  // 128^-0.5
#define EPS_C   1e-6f
#define LOG2_THETA 19.931568569324174f // log2(1e6)

typedef __attribute__((ext_vector_type(16))) _Float16 v16h;
typedef __attribute__((ext_vector_type(8)))  _Float16 v8h;
typedef __attribute__((ext_vector_type(8)))  float    v8f;

// ---------------------------------------------------------------------------
// WMMA helpers (CDNA5 V_WMMA_F32_16X16X32_F16, fp32 accumulate)
// ---------------------------------------------------------------------------
__device__ __forceinline__ v8f wmma_f16(v16h a, v16h b, v8f c) {
    // 8 args: (neg_a, A, neg_b, B, c_mod, C, reuse_a, reuse_b)
    return __builtin_amdgcn_wmma_f32_16x16x32_f16(false, a, false, b,
                                                  (short)0, c, false, false);
}

// A-fragment (16x32 f16, MxK): lane owns row m = lane&15.
// low lanes:  a[0..15] = K 0..15   (koff = 0)
// high lanes: a[0..15] = K 16..31  (koff = 16)
__device__ __forceinline__ v16h load_a_frag(const _Float16* __restrict__ rowptr,
                                            int koff) {
    v8h lo = *(const v8h*)(rowptr + koff);
    v8h hi = *(const v8h*)(rowptr + koff + 8);
    v16h a;
#pragma unroll
    for (int i = 0; i < 8; ++i) { a[i] = lo[i]; a[i + 8] = hi[i]; }
    return a;
}

// ---------------------------------------------------------------------------
// 1) fp32 -> f16 elementwise convert
// ---------------------------------------------------------------------------
__global__ __launch_bounds__(256)
void cvt_f16_kernel(const float* __restrict__ X, _Float16* __restrict__ Y,
                    size_t n) {
    size_t i = (size_t)blockIdx.x * 256 + threadIdx.x;
    if (i < n) Y[i] = (_Float16)X[i];
}

// ---------------------------------------------------------------------------
// 2) fp32 W[K][N] -> f16 Wt[N][K]  (LDS-tiled transpose, 32x32 tiles)
// ---------------------------------------------------------------------------
__global__ __launch_bounds__(256)
void transpose_cvt_kernel(const float* __restrict__ W, _Float16* __restrict__ Wt,
                          int K, int N) {
    __shared__ float tile[32][33];
    const int x = threadIdx.x & 31;
    const int y = threadIdx.x >> 5;      // 0..7
    const int kb = blockIdx.y * 32;
    const int nb = blockIdx.x * 32;
#pragma unroll
    for (int i = 0; i < 4; ++i)
        tile[y + 8 * i][x] = W[(size_t)(kb + y + 8 * i) * N + nb + x];
    __syncthreads();
#pragma unroll
    for (int i = 0; i < 4; ++i)
        Wt[(size_t)(nb + y + 8 * i) * K + kb + x] = (_Float16)tile[x][y + 8 * i];
}

// ---------------------------------------------------------------------------
// 3) Generic WMMA GEMM: C(f32, MxN) = A(f16, MxK row-major) * Bt(f16, NxK)^T
//    Block = 128 threads (4 waves). Block tile 128x64; wave tile 32x64
//    (2 A-fragments x 4 B-fragments -> 8 WMMAs per 32-deep k-step,
//     ~21 FLOP per operand byte fetched).
// ---------------------------------------------------------------------------
__global__ __launch_bounds__(128)
void gemm_bt_kernel(const _Float16* __restrict__ A,
                    const _Float16* __restrict__ Bt,
                    float* __restrict__ C, int M, int N, int K) {
    const int lane   = threadIdx.x & 31;
    const int wave   = threadIdx.x >> 5;
    const int laneHi = lane >> 4;
    const int lane15 = lane & 15;
    const int m0 = blockIdx.y * 128 + wave * 32;
    const int n0 = blockIdx.x * 64;

    v8f acc[2][4] = {};
    const _Float16* arow0 = A + (size_t)(m0 + lane15) * K;
    const _Float16* arow1 = arow0 + (size_t)16 * K;
    const int ka = laneHi ? 16 : 0;    // A-fragment K offset
    const int kb = laneHi ? 16 : 0;    // B-fragment K offset

    for (int k0 = 0; k0 < K; k0 += 32) {
        __builtin_prefetch(arow0 + k0 + 128, 0, 1);  // global_prefetch next lines
        __builtin_prefetch(arow1 + k0 + 128, 0, 1);
        v16h a0 = load_a_frag(arow0, k0 + ka);
        v16h a1 = load_a_frag(arow1, k0 + ka);
#pragma unroll
        for (int nt = 0; nt < 4; ++nt) {
            const _Float16* bp = Bt + (size_t)(n0 + nt * 16 + lane15) * K + k0 + kb;
            v16h bfrag = *(const v16h*)bp;
            acc[0][nt] = wmma_f16(a0, bfrag, acc[0][nt]);
            acc[1][nt] = wmma_f16(a1, bfrag, acc[1][nt]);
        }
    }
    // C/D layout: VGPR r -> row m = r (+8 for high lanes), col n = lane&15
#pragma unroll
    for (int mt = 0; mt < 2; ++mt) {
#pragma unroll
        for (int nt = 0; nt < 4; ++nt) {
#pragma unroll
            for (int r = 0; r < 8; ++r) {
                int row = m0 + mt * 16 + (laneHi ? 8 : 0) + r;
                int col = n0 + nt * 16 + lane15;
                C[(size_t)row * N + col] = acc[mt][nt][r];
            }
        }
    }
}

// ---------------------------------------------------------------------------
// 4) Fused RMSNorm + RoPE (f32 in [tok][H*128] -> f16 out [b][h][s][128])
//    One wave per (token, head) row of 128. 4 waves per block.
// ---------------------------------------------------------------------------
__global__ __launch_bounds__(128)
void norm_rope_kernel(const float* __restrict__ X, const int* __restrict__ pos,
                      const float* __restrict__ w, _Float16* __restrict__ Out,
                      int H) {
    const int lane = threadIdx.x & 31;
    const int wave = threadIdx.x >> 5;
    const int rid  = blockIdx.x * 4 + wave;     // (b*S + s)*H + h
    const int h    = rid % H;
    const int tok  = rid / H;                   // b*S + s
    const int b    = tok / S_C;
    const int s    = tok % S_C;

    const float* xin = X + (size_t)rid * D_C;
    const int j = lane * 2;
    float x1a = xin[j],      x1b = xin[j + 1];
    float x2a = xin[64 + j], x2b = xin[64 + j + 1];

    float ss = x1a * x1a + x1b * x1b + x2a * x2a + x2b * x2b;
#pragma unroll
    for (int off = 16; off > 0; off >>= 1) ss += __shfl_xor(ss, off, 32);
    const float rinv = rsqrtf(ss * (1.0f / (float)D_C) + EPS_C);

    x1a *= rinv * w[j];      x1b *= rinv * w[j + 1];
    x2a *= rinv * w[64 + j]; x2b *= rinv * w[64 + j + 1];

    const float p  = (float)pos[tok];
    const float fa = p * exp2f(-(float)j * (LOG2_THETA / 64.0f));
    const float fb = p * exp2f(-(float)(j + 1) * (LOG2_THETA / 64.0f));
    const float ca = cosf(fa), sa = sinf(fa);
    const float cb = cosf(fb), sb = sinf(fb);

    _Float16* orow = Out + ((size_t)(b * H + h) * S_C + s) * D_C;
    orow[j]          = (_Float16)(x1a * ca - x2a * sa);
    orow[j + 1]      = (_Float16)(x1b * cb - x2b * sb);
    orow[64 + j]     = (_Float16)(x2a * ca + x1a * sa);
    orow[64 + j + 1] = (_Float16)(x2b * cb + x1b * sb);
}

// ---------------------------------------------------------------------------
// 5) V: f32 [tok][NKV*128] -> f16 Vt [b][hk][d][s]  (so P·V B-operand is
//    contiguous along the KV axis)
// ---------------------------------------------------------------------------
__global__ __launch_bounds__(256)
void v_transpose_kernel(const float* __restrict__ V, _Float16* __restrict__ Vt) {
    size_t idx = (size_t)blockIdx.x * 256 + threadIdx.x;   // over TOK*NKV*D
    const int d = (int)(idx & (D_C - 1));
    size_t t = idx >> 7;
    const int h = (int)(t & (NKV_C - 1));
    size_t tok = t >> 2;
    const int s = (int)(tok % S_C);
    const int b = (int)(tok / S_C);
    Vt[((size_t)(b * NKV_C + h) * D_C + d) * S_C + s] = (_Float16)V[idx];
}

// ---------------------------------------------------------------------------
// 6) Flash attention. One wave per (b, h, 16-token Q tile).
//    KV chunks of 32; online softmax; all matmuls via WMMA f16.
//    Qh: [b][h][s][128] f16, Kh: [b][hk][s][128] f16, Vt: [b][hk][d][s] f16
//    Ctx out: f16 [tok][NH*128]
// ---------------------------------------------------------------------------
__global__ __launch_bounds__(32)
void flash_attn_kernel(const _Float16* __restrict__ Qh,
                       const _Float16* __restrict__ Kh,
                       const _Float16* __restrict__ Vt,
                       _Float16* __restrict__ Ctx) {
    __shared__ _Float16 Pld[16 * 32];
    const int lane   = threadIdx.x & 31;
    const int laneHi = lane >> 4;
    const int lane15 = lane & 15;

    int bid = blockIdx.x;                       // b*NH*(S/16) + h*(S/16) + qt
    const int qt = bid % (S_C / 16); bid /= (S_C / 16);
    const int h  = bid % NH_C;
    const int b  = bid / NH_C;
    const int hk = h >> 2;                      // NH/NKV = 4
    const int q0 = qt * 16;

    const int ka   = laneHi ? 16 : 0;
    const int kb16 = laneHi ? 16 : 0;

    // Q tile A-fragments for all 4 K-chunks of D=128 (kept in registers)
    const _Float16* qbase =
        Qh + ((size_t)(b * NH_C + h) * S_C + q0 + lane15) * D_C;
    v16h qf[4];
#pragma unroll
    for (int ks = 0; ks < 4; ++ks) qf[ks] = load_a_frag(qbase, ks * 32 + ka);

    const _Float16* kbase = Kh + (size_t)(b * NKV_C + hk) * S_C * D_C;
    const _Float16* vbase = Vt + (size_t)(b * NKV_C + hk) * D_C * S_C;

    v8f acc[8] = {};
    float mrow[8], lrow[8];
#pragma unroll
    for (int r = 0; r < 8; ++r) { mrow[r] = -1e30f; lrow[r] = 0.0f; }

    for (int kv0 = 0; kv0 < S_C; kv0 += 32) {
        // ---- scores: S(16x32) = Q(16x128) . K_chunk(32x128)^T --------------
        v8f s0 = {}, s1 = {};
#pragma unroll
        for (int ks = 0; ks < 4; ++ks) {
            const _Float16* bp0 =
                kbase + (size_t)(kv0 + lane15) * D_C + ks * 32 + kb16;
            const _Float16* bp1 = bp0 + 16 * D_C;
            s0 = wmma_f16(qf[ks], *(const v16h*)bp0, s0);
            s1 = wmma_f16(qf[ks], *(const v16h*)bp1, s1);
        }
        // ---- online softmax (row stats per 16-lane half) -------------------
        float p0[8], p1[8], alpha[8];
#pragma unroll
        for (int r = 0; r < 8; ++r) {
            float sc0 = s0[r] * SCALE_C;
            float sc1 = s1[r] * SCALE_C;
            float mx = fmaxf(sc0, sc1);
#pragma unroll
            for (int off = 8; off > 0; off >>= 1)
                mx = fmaxf(mx, __shfl_xor(mx, off, 32));
            float mnew = fmaxf(mrow[r], mx);
            alpha[r] = __expf(mrow[r] - mnew);
            p0[r]    = __expf(sc0 - mnew);
            p1[r]    = __expf(sc1 - mnew);
            float t = p0[r] + p1[r];
#pragma unroll
            for (int off = 8; off > 0; off >>= 1) t += __shfl_xor(t, off, 32);
            lrow[r] = lrow[r] * alpha[r] + t;
            mrow[r] = mnew;
        }
#pragma unroll
        for (int nt = 0; nt < 8; ++nt)
#pragma unroll
            for (int r = 0; r < 8; ++r) acc[nt][r] *= alpha[r];

        // ---- C-layout -> A-layout transpose of P via LDS -------------------
#pragma unroll
        for (int r = 0; r < 8; ++r) {
            int row = (laneHi ? 8 : 0) + r;
            Pld[row * 32 + lane15]      = (_Float16)p0[r];
            Pld[row * 32 + 16 + lane15] = (_Float16)p1[r];
        }
        __syncthreads();
        v16h pa;
        {
            const _Float16* pr = &Pld[lane15 * 32 + ka];
#pragma unroll
            for (int i = 0; i < 8; ++i) { pa[i] = pr[i]; pa[i + 8] = pr[8 + i]; }
        }
        __syncthreads();

        // ---- ctx += P(16x32) . V_chunk(32x128) -----------------------------
#pragma unroll
        for (int nt = 0; nt < 8; ++nt) {
            const _Float16* vp =
                vbase + (size_t)(nt * 16 + lane15) * S_C + kv0 + kb16;
            acc[nt] = wmma_f16(pa, *(const v16h*)vp, acc[nt]);
        }
    }

    // ---- normalize and write ctx (f16, [tok][h*128+d]) ---------------------
#pragma unroll
    for (int nt = 0; nt < 8; ++nt) {
#pragma unroll
        for (int r = 0; r < 8; ++r) {
            int row = q0 + (laneHi ? 8 : 0) + r;
            int col = h * D_C + nt * 16 + lane15;
            Ctx[(size_t)(b * S_C + row) * (NH_C * D_C) + col] =
                (_Float16)(acc[nt][r] / lrow[r]);
        }
    }
}

// ---------------------------------------------------------------------------
// Launch
// ---------------------------------------------------------------------------
extern "C" void kernel_launch(void* const* d_in, const int* in_sizes, int n_in,
                              void* d_out, int out_size, void* d_ws, size_t ws_size,
                              hipStream_t stream) {
    const float* Xf  = (const float*)d_in[0];
    const int*   pos = (const int*)d_in[1];
    const float* Wq  = (const float*)d_in[2];
    const float* Wk  = (const float*)d_in[3];
    const float* Wv  = (const float*)d_in[4];
    const float* Wo  = (const float*)d_in[5];
    const float* qw  = (const float*)d_in[6];
    const float* kw  = (const float*)d_in[7];
    float* out = (float*)d_out;

    char* ws = (char*)d_ws;
    size_t off = 0;
    auto carve = [&](size_t bytes) -> void* {
        off = (off + 255) & ~(size_t)255;
        void* p = ws + off;
        off += bytes;
        return p;
    };

    const size_t TOK = TOK_C;
    _Float16* Xh   = (_Float16*)carve(TOK * HID_C * 2);          // 16.8 MB
    _Float16* Wq_t = (_Float16*)carve((size_t)HID_C * (NH_C * D_C) * 2);
    _Float16* Wk_t = (_Float16*)carve((size_t)HID_C * (NKV_C * D_C) * 2);
    _Float16* Wv_t = (_Float16*)carve((size_t)HID_C * (NKV_C * D_C) * 2);
    _Float16* Wo_t = (_Float16*)carve((size_t)(NH_C * D_C) * HID_C * 2);
    _Float16* Qh   = (_Float16*)carve(TOK * (size_t)(NH_C * D_C) * 2);
    _Float16* Kh   = (_Float16*)carve(TOK * (size_t)(NKV_C * D_C) * 2);
    _Float16* Vt   = (_Float16*)carve(TOK * (size_t)(NKV_C * D_C) * 2);
    float*    Qf32 = (float*)carve(TOK * (size_t)(NH_C * D_C) * 4);   // 33.5 MB
    float*    Kf32 = (float*)carve(TOK * (size_t)(NKV_C * D_C) * 4);
    float*    Vf32 = (float*)carve(TOK * (size_t)(NKV_C * D_C) * 4);
    // Ctx (f16, 16.8 MB) aliases Qf32 (dead after norm_rope on Q)
    _Float16* Ctxh = (_Float16*)Qf32;

    // 1) convert activations to f16
    {
        size_t n = TOK * HID_C;
        cvt_f16_kernel<<<dim3((unsigned)((n + 255) / 256)), 256, 0, stream>>>(Xf, Xh, n);
    }
    // 2) transpose-convert weights: W[K][N] -> Wt[N][K]
    transpose_cvt_kernel<<<dim3(2048 / 32, 2048 / 32), 256, 0, stream>>>(Wq, Wq_t, HID_C, NH_C * D_C);
    transpose_cvt_kernel<<<dim3(512 / 32, 2048 / 32), 256, 0, stream>>>(Wk, Wk_t, HID_C, NKV_C * D_C);
    transpose_cvt_kernel<<<dim3(512 / 32, 2048 / 32), 256, 0, stream>>>(Wv, Wv_t, HID_C, NKV_C * D_C);
    transpose_cvt_kernel<<<dim3(2048 / 32, 2048 / 32), 256, 0, stream>>>(Wo, Wo_t, NH_C * D_C, HID_C);

    // 3) QKV projections (WMMA GEMMs, fp32 out). Block tile 128x64.
    gemm_bt_kernel<<<dim3((NH_C * D_C) / 64, TOK_C / 128), 128, 0, stream>>>(Xh, Wq_t, Qf32, TOK_C, NH_C * D_C, HID_C);
    gemm_bt_kernel<<<dim3((NKV_C * D_C) / 64, TOK_C / 128), 128, 0, stream>>>(Xh, Wk_t, Kf32, TOK_C, NKV_C * D_C, HID_C);
    gemm_bt_kernel<<<dim3((NKV_C * D_C) / 64, TOK_C / 128), 128, 0, stream>>>(Xh, Wv_t, Vf32, TOK_C, NKV_C * D_C, HID_C);

    // 4) RMSNorm + RoPE -> f16, head-major layouts
    norm_rope_kernel<<<dim3(TOK_C * NH_C / 4), 128, 0, stream>>>(Qf32, pos, qw, Qh, NH_C);
    norm_rope_kernel<<<dim3(TOK_C * NKV_C / 4), 128, 0, stream>>>(Kf32, pos, kw, Kh, NKV_C);

    // 5) V -> f16, transposed per head: [b][hk][d][s]
    {
        size_t n = TOK * (size_t)(NKV_C * D_C);
        v_transpose_kernel<<<dim3((unsigned)(n / 256)), 256, 0, stream>>>(Vf32, Vt);
    }

    // 6) flash attention -> ctx f16 [tok][NH*D]
    flash_attn_kernel<<<dim3(B_C * NH_C * (S_C / 16)), 32, 0, stream>>>(Qh, Kh, Vt, Ctxh);

    // 7) output projection -> fp32 d_out
    gemm_bt_kernel<<<dim3(HID_C / 64, TOK_C / 128), 128, 0, stream>>>(Ctxh, Wo_t, out, TOK_C, HID_C, NH_C * D_C);

    (void)in_sizes; (void)n_in; (void)out_size; (void)ws_size;
}
